// CrossAttention_40372692582970
// MI455X (gfx1250) — matrix operand order repaired and checked by reading, not compiled, not run on hardware
//
#include <hip/hip_runtime.h>
#include <hip/hip_bf16.h>

typedef __attribute__((ext_vector_type(16))) _Float16 half16;
typedef __attribute__((ext_vector_type(8)))  float    float8;
typedef int v4i_vs __attribute__((vector_size(4 * sizeof(int))));

#define AS1 __attribute__((address_space(1)))
#define AS3 __attribute__((address_space(3)))

#if defined(__has_builtin)
#if __has_builtin(__builtin_amdgcn_global_load_async_to_lds_b128)
#define HAVE_ASYNC_LDS 1
#endif
#endif
#ifndef HAVE_ASYNC_LDS
#define HAVE_ASYNC_LDS 0
#endif

#if HAVE_ASYNC_LDS
__device__ __forceinline__ void async_copy16(const _Float16* g, _Float16* lds) {
    __builtin_amdgcn_global_load_async_to_lds_b128(
        (AS1 v4i_vs*)g, (AS3 v4i_vs*)lds, 0, 0);
}
#endif

__device__ __forceinline__ void wait_async0() {
#if HAVE_ASYNC_LDS
#if __has_builtin(__builtin_amdgcn_s_wait_asynccnt)
    __builtin_amdgcn_s_wait_asynccnt(0);
#else
    asm volatile("s_wait_asynccnt 0x0" ::: "memory");
#endif
#endif
}

// DPP16 row_ror butterfly: full reduction across a 16-lane row (VALU only,
// no LDS traffic — half-wave rows match the WMMA C-tile row placement).
template <int CTRL>
__device__ __forceinline__ float dpp_rot(float x) {
    return __builtin_bit_cast(float,
        __builtin_amdgcn_mov_dpp(__builtin_bit_cast(int, x), CTRL, 0xf, 0xf, true));
}
__device__ __forceinline__ float row_max16(float v) {
    v = fmaxf(v, dpp_rot<0x128>(v));   // row_ror:8
    v = fmaxf(v, dpp_rot<0x124>(v));   // row_ror:4
    v = fmaxf(v, dpp_rot<0x122>(v));   // row_ror:2
    v = fmaxf(v, dpp_rot<0x121>(v));   // row_ror:1
    return v;
}
__device__ __forceinline__ float row_sum16(float v) {
    v += dpp_rot<0x128>(v);
    v += dpp_rot<0x124>(v);
    v += dpp_rot<0x122>(v);
    v += dpp_rot<0x121>(v);
    return v;
}

// ---------------------------------------------------------------------------
// Tiled GEMM: C[M,N] = A[M,K] * B[K,N] (+bias), f16 WMMA with f32 accumulate.
// Block = 256 threads (8 waves), tile 128(M) x 64(N), BK = 32.
// f16 A tiles are staged with GLOBAL_LOAD_ASYNC_TO_LDS_B128 when available.
// ---------------------------------------------------------------------------
template <typename AT, typename OT, bool BIAS>
__global__ __launch_bounds__(256) void gemm_f16_wmma(
    const AT* __restrict__ A, const float* __restrict__ Bw,
    const float* __restrict__ bias, OT* __restrict__ C,
    int M, int N, int K)
{
    constexpr int BM = 128, BN = 64, BK = 32;
    constexpr int ASTR = BK + 16;   // 48 halfs = 96B rows (16B aligned)
    constexpr int BSTR = BK + 16;
    constexpr bool A_IS_F16 = __is_same(AT, _Float16);
    __shared__ _Float16 As[BM][ASTR];     // [m][k]
    __shared__ _Float16 Bst[BN][BSTR];    // transposed: [n][k]

    const int tid  = threadIdx.x;
    const int lane = tid & 31;
    const int wave = tid >> 5;
    const int m0 = blockIdx.y * BM;
    const int n0 = blockIdx.x * BN;

    const int row16 = lane & 15;
    const int hi    = lane >> 4;     // lane half
    const int kbA   = hi * 16;       // A-frag K base (contiguous, matches B)
    const int kbB   = hi * 16;       // B-frag K base (contiguous, matches A)

    float8 acc[4] = {};

    for (int k0 = 0; k0 < K; k0 += BK) {
        // Stage A tile
        if constexpr (A_IS_F16) {
#if HAVE_ASYNC_LDS
            // 128 rows x 32 halfs = 512 x 16B chunks; 256 lanes x 2 issues
            #pragma unroll
            for (int it = 0; it < 2; ++it) {
                int chunk = it * 256 + tid;
                int r = chunk >> 2, c = (chunk & 3) * 8;
                async_copy16((const _Float16*)A + (size_t)(m0 + r) * K + (k0 + c),
                             &As[r][c]);
            }
#else
            for (int idx = tid; idx < BM * BK; idx += 256) {
                int r = idx >> 5, c = idx & 31;
                As[r][c] = A[(size_t)(m0 + r) * K + (k0 + c)];
            }
#endif
        } else {
            for (int idx = tid; idx < BM * BK; idx += 256) {
                int r = idx >> 5, c = idx & 31;
                As[r][c] = (_Float16)A[(size_t)(m0 + r) * K + (k0 + c)];
            }
        }
        // Stage B tile transposed (f32 -> f16 conversion, VALU path)
        for (int idx = tid; idx < BK * BN; idx += 256) {
            int r = idx >> 6, c = idx & 63;
            Bst[c][r] = (_Float16)Bw[(size_t)(k0 + r) * N + (n0 + c)];
        }
        if constexpr (A_IS_F16) wait_async0();
        __syncthreads();

        // A fragment: one 16x32 tile per wave (contiguous -> ds_load_b128)
        half16 a;
        {
            const _Float16* ap = &As[wave * 16 + row16][0];
            #pragma unroll
            for (int i = 0; i < 8; ++i) {
                a[i]     = ap[kbA + i];          // K = kbA..kbA+7
                a[8 + i] = ap[kbA + 8 + i];      // K = kbA+8..kbA+15
            }
        }
        #pragma unroll
        for (int t = 0; t < 4; ++t) {
            half16 b;
            const _Float16* bp = &Bst[t * 16 + row16][0];
            #pragma unroll
            for (int i = 0; i < 16; ++i) b[i] = bp[kbB + i];
            acc[t] = __builtin_amdgcn_wmma_f32_16x16x32_f16(
                false, a, false, b, (short)0, acc[t], false, false);
        }
        __syncthreads();
    }

    // Epilogue: C/D layout — lane half hi, VGPR j -> row hi*8+j, col = lane&15
    #pragma unroll
    for (int t = 0; t < 4; ++t) {
        #pragma unroll
        for (int j = 0; j < 8; ++j) {
            int rowm = m0 + wave * 16 + hi * 8 + j;
            int coln = n0 + t * 16 + row16;
            float v = acc[t][j];
            if constexpr (BIAS) v += bias[coln];
            C[(size_t)rowm * N + coln] = (OT)v;
        }
    }
}

// ---------------------------------------------------------------------------
// Flash attention per (batch, head, 64-row query tile). 128 threads = 4 waves,
// each wave owns 16 query rows. Online softmax; all matmuls via WMMA.
// Q/K tiles staged with async-to-LDS; V staged via VGPRs (needs transpose).
// ---------------------------------------------------------------------------
__global__ __launch_bounds__(128) void flash_attn_wmma(
    const _Float16* __restrict__ Q, const _Float16* __restrict__ Kp,
    const _Float16* __restrict__ Vp, _Float16* __restrict__ Oa)
{
    constexpr int D = 64, TQ = 64, TK = 64;
    constexpr int NQ = 2048, NKEY = 1024, HD = 1024;
    constexpr int QSTR = D + 8;    // 72 halfs = 144B (16B aligned)
    __shared__ _Float16 Qs[TQ][QSTR];        // [q][d]
    __shared__ _Float16 Ks[TK][QSTR];        // [key][d]
    __shared__ _Float16 Vt[D][TK + 8];       // transposed: [d][key]
    __shared__ _Float16 Ps[TQ][TK + 8];      // P tile, per-wave 16-row bands

    const int tid  = threadIdx.x;
    const int lane = tid & 31;
    const int wave = tid >> 5;
    const int b  = blockIdx.z;
    const int h  = blockIdx.y;
    const int qt = blockIdx.x;

    const int row16 = lane & 15;
    const int hi    = lane >> 4;
    const int kbA   = hi * 16;
    const int kbB   = hi * 16;

    const size_t qbase  = ((size_t)b * NQ + qt * TQ) * HD + h * D;
    const size_t kvbase = ((size_t)b * NKEY) * HD + h * D;

    // Load Q tile (async: drained by the wait before the first barrier)
#if HAVE_ASYNC_LDS
    #pragma unroll
    for (int it = 0; it < 4; ++it) {
        int chunk = it * 128 + tid;              // 64 rows x 8 chunks of 16B
        int r = chunk >> 3, c = (chunk & 7) * 8;
        async_copy16(Q + qbase + (size_t)r * HD + c, &Qs[r][c]);
    }
#else
    for (int idx = tid; idx < TQ * D; idx += 128) {
        int r = idx >> 6, c = idx & 63;
        Qs[r][c] = Q[qbase + (size_t)r * HD + c];
    }
#endif

    float m[8], l[8];
    float8 o[4] = {};
    #pragma unroll
    for (int j = 0; j < 8; ++j) { m[j] = -__builtin_inff(); l[j] = 0.f; }

    for (int kt = 0; kt < NKEY / TK; ++kt) {
        // Stage K tile (async) and transposed V tile (VALU)
#if HAVE_ASYNC_LDS
        #pragma unroll
        for (int it = 0; it < 4; ++it) {
            int chunk = it * 128 + tid;
            int r = chunk >> 3, c = (chunk & 7) * 8;
            async_copy16(Kp + kvbase + (size_t)(kt * TK + r) * HD + c, &Ks[r][c]);
        }
        for (int idx = tid; idx < TK * D; idx += 128) {
            int r = idx >> 6, c = idx & 63;
            Vt[c][r] = Vp[kvbase + (size_t)(kt * TK + r) * HD + c];
        }
#else
        for (int idx = tid; idx < TK * D; idx += 128) {
            int r = idx >> 6, c = idx & 63;
            size_t g = kvbase + (size_t)(kt * TK + r) * HD + c;
            Ks[r][c] = Kp[g];
            Vt[c][r] = Vp[g];
        }
#endif
        // Prefetch next K/V tile into caches (global_prefetch_b8)
        if (kt + 1 < NKEY / TK) {
            int pr = tid & 63, pc = (tid >> 6) * 32;
            size_t g = kvbase + (size_t)((kt + 1) * TK + pr) * HD + pc;
            __builtin_prefetch(Kp + g, 0, 0);
            __builtin_prefetch(Vp + g, 0, 0);
        }
        wait_async0();
        __syncthreads();

        // S = Q * K^T  (B-frag element (k=d, n=key) = Ks[key][d]: contiguous)
        float8 s[4] = {};
        #pragma unroll
        for (int kc = 0; kc < 2; ++kc) {
            half16 a;
            const _Float16* ap = &Qs[wave * 16 + row16][kc * 32];
            #pragma unroll
            for (int i = 0; i < 8; ++i) {
                a[i] = ap[kbA + i]; a[8 + i] = ap[kbA + 8 + i];
            }
            #pragma unroll
            for (int nt = 0; nt < 4; ++nt) {
                half16 bf;
                const _Float16* bp = &Ks[nt * 16 + row16][kc * 32];
                #pragma unroll
                for (int i = 0; i < 16; ++i) bf[i] = bp[kbB + i];
                s[nt] = __builtin_amdgcn_wmma_f32_16x16x32_f16(
                    false, a, false, bf, (short)0, s[nt], false, false);
            }
        }

        // Online softmax. Row r of the C tile sits in VGPR slot (r&7) of one
        // lane-half; row reductions use DPP row_ror butterflies (VALU only).
        const float scale = 0.125f;   // 64^-0.5
        #pragma unroll
        for (int j = 0; j < 8; ++j) {
            float rmax = -__builtin_inff();
            #pragma unroll
            for (int nt = 0; nt < 4; ++nt) {
                s[nt][j] *= scale;
                rmax = fmaxf(rmax, s[nt][j]);
            }
            rmax = row_max16(rmax);
            float mn = fmaxf(m[j], rmax);
            float al = __expf(m[j] - mn);
            float rsum = 0.f;
            #pragma unroll
            for (int nt = 0; nt < 4; ++nt) {
                float p = __expf(s[nt][j] - mn);
                s[nt][j] = p;
                rsum += p;
            }
            rsum = row_sum16(rsum);
            l[j] = l[j] * al + rsum;
            m[j] = mn;
            #pragma unroll
            for (int t = 0; t < 4; ++t) o[t][j] *= al;
        }

        // C-layout -> A-fragment reshape through LDS (per-wave band, no barrier)
        #pragma unroll
        for (int nt = 0; nt < 4; ++nt)
            #pragma unroll
            for (int j = 0; j < 8; ++j)
                Ps[wave * 16 + hi * 8 + j][nt * 16 + row16] = (_Float16)s[nt][j];

        // O += P * V   (B-frag element (k=key, n=d) = Vt[d][key]: contiguous)
        #pragma unroll
        for (int kc = 0; kc < 2; ++kc) {
            half16 a;
            const _Float16* ap = &Ps[wave * 16 + row16][kc * 32];
            #pragma unroll
            for (int i = 0; i < 8; ++i) {
                a[i] = ap[kbA + i]; a[8 + i] = ap[kbA + 8 + i];
            }
            #pragma unroll
            for (int t = 0; t < 4; ++t) {
                half16 bf;
                const _Float16* bp = &Vt[t * 16 + row16][kc * 32];
                #pragma unroll
                for (int i = 0; i < 16; ++i) bf[i] = bp[kbB + i];
                o[t] = __builtin_amdgcn_wmma_f32_16x16x32_f16(
                    false, a, false, bf, (short)0, o[t], false, false);
            }
        }
        __syncthreads();
    }

    // Finalize: divide by softmax denom, store f16 [b, n, h*64+d]
    #pragma unroll
    for (int j = 0; j < 8; ++j) {
        float inv = 1.0f / l[j];
        int rq = qt * TQ + wave * 16 + hi * 8 + j;
        #pragma unroll
        for (int t = 0; t < 4; ++t) {
            Oa[((size_t)b * NQ + rq) * HD + h * D + t * 16 + row16] =
                (_Float16)(o[t][j] * inv);
        }
    }
}

// ---------------------------------------------------------------------------
extern "C" void kernel_launch(void* const* d_in, const int* in_sizes, int n_in,
                              void* d_out, int out_size, void* d_ws, size_t ws_size,
                              hipStream_t stream) {
    (void)in_sizes; (void)n_in; (void)out_size; (void)ws_size;
    const float* x   = (const float*)d_in[0];   // [2,2048,1024]
    const float* ctx = (const float*)d_in[1];   // [2,1024,768]
    const float* Wq  = (const float*)d_in[2];   // [1024,1024]
    const float* Wk  = (const float*)d_in[3];   // [768,1024]
    const float* Wv  = (const float*)d_in[4];   // [768,1024]
    const float* Wo  = (const float*)d_in[5];   // [1024,1024]
    const float* bo  = (const float*)d_in[6];   // [1024]
    float* out = (float*)d_out;                 // [2,2048,1024]

    // f16 workspace: Q(8MB) K(4MB) V(4MB) attn(8MB)
    _Float16* qws = (_Float16*)d_ws;
    _Float16* kws = qws + (size_t)4096 * 1024;
    _Float16* vws = kws + (size_t)2048 * 1024;
    _Float16* aws = vws + (size_t)2048 * 1024;

    // Projections: Q = x*Wq, K = ctx*Wk, V = ctx*Wv (f16 out)
    gemm_f16_wmma<float, _Float16, false>
        <<<dim3(1024 / 64, 4096 / 128), 256, 0, stream>>>(x, Wq, nullptr, qws, 4096, 1024, 1024);
    gemm_f16_wmma<float, _Float16, false>
        <<<dim3(1024 / 64, 2048 / 128), 256, 0, stream>>>(ctx, Wk, nullptr, kws, 2048, 1024, 768);
    gemm_f16_wmma<float, _Float16, false>
        <<<dim3(1024 / 64, 2048 / 128), 256, 0, stream>>>(ctx, Wv, nullptr, vws, 2048, 1024, 768);

    // Attention: grid (query tiles, heads, batch)
    flash_attn_wmma<<<dim3(2048 / 64, 16, 2), 128, 0, stream>>>(qws, kws, vws, aws);

    // Output projection with bias (f32 out)
    gemm_f16_wmma<_Float16, float, true>
        <<<dim3(1024 / 64, 4096 / 128), 256, 0, stream>>>(aws, Wo, bo, out, 4096, 1024, 1024);
}
